// SingleModel_86577950753154
// MI455X (gfx1250) — matrix operand order repaired and checked by reading, not compiled, not run on hardware
//
#include <hip/hip_runtime.h>
#include <hip/hip_bf16.h>

typedef __attribute__((ext_vector_type(16))) __bf16 v16bf;
typedef __attribute__((ext_vector_type(8)))  __bf16 v8bf;
typedef __attribute__((ext_vector_type(4)))  __bf16 v4bf;
typedef __attribute__((ext_vector_type(8)))  float  v8f;

#define INIT_DIM 512
#define HID      1024
#define OUTD     64
#define N_NBR    65536
#define ROWS_WG  128
#define LDS_STRIDE 520          // 512 + 8 bf16 pad to break bank alignment
#define W_ELEMS  (HID * INIT_DIM)   // 524288 per relation type

// ---------------------------------------------------------------------------
// Kernel 1: convert W1_{a,b,c} f32 -> bf16 into workspace; zero global colsum
// ---------------------------------------------------------------------------
__global__ void k_convert(const float* __restrict__ Wa,
                          const float* __restrict__ Wb,
                          const float* __restrict__ Wc,
                          __bf16* __restrict__ Wout,
                          float* __restrict__ colsum) {
    const int t    = blockIdx.x * blockDim.x + threadIdx.x; // 0..131071
    const int type = blockIdx.y;
    const float* W = (type == 0) ? Wa : ((type == 1) ? Wb : Wc);
    float4 v = ((const float4*)W)[t];
    v4bf o;
    o[0] = (__bf16)v.x; o[1] = (__bf16)v.y; o[2] = (__bf16)v.z; o[3] = (__bf16)v.w;
    ((v4bf*)(Wout + (size_t)type * W_ELEMS))[t] = o;
    if (type == 0 && t < HID) colsum[t] = 0.0f;
}

// ---------------------------------------------------------------------------
// Kernel 2: fused gather + bf16 WMMA GEMM + bias + ReLU + column-sum
// grid = (N_NBR/ROWS_WG, 3), block = 256 (8 waves)
// ---------------------------------------------------------------------------
__global__ void __launch_bounds__(256, 1)
k_main(const float* __restrict__ af_a, const float* __restrict__ af_b,
       const float* __restrict__ af_c,
       const int* __restrict__ nb_a, const int* __restrict__ nb_b,
       const int* __restrict__ nb_c,
       const float* __restrict__ b1_a, const float* __restrict__ b1_b,
       const float* __restrict__ b1_c,
       const __bf16* __restrict__ Wbf,
       float* __restrict__ colsum) {
    __shared__ __bf16 Atile[ROWS_WG * LDS_STRIDE];  // 133120 B
    __shared__ float  csum[HID];                     // 4096 B

    const int type = blockIdx.y;
    const float*  af = (type == 0) ? af_a : ((type == 1) ? af_b : af_c);
    const int*    nb = (type == 0) ? nb_a : ((type == 1) ? nb_b : nb_c);
    const float*  b1 = (type == 0) ? b1_a : ((type == 1) ? b1_b : b1_c);
    const __bf16* W  = Wbf + (size_t)type * W_ELEMS;

    const int tid = threadIdx.x;
    for (int i = tid; i < HID; i += 256) csum[i] = 0.0f;

    // ---- stage 128 gathered rows (f32 -> bf16) into LDS: 2 threads/row ----
    {
        const int row = tid >> 1;
        const int c0  = (tid & 1) * 256;
        const int g   = nb[blockIdx.x * ROWS_WG + row];
        const float4* src = (const float4*)(af + (size_t)g * INIT_DIM + c0);
        __bf16* dst = &Atile[row * LDS_STRIDE + c0];
        #pragma unroll
        for (int i = 0; i < 64; ++i) {
            float4 v = src[i];
            v4bf o;
            o[0] = (__bf16)v.x; o[1] = (__bf16)v.y;
            o[2] = (__bf16)v.z; o[3] = (__bf16)v.w;
            *(v4bf*)(dst + i * 4) = o;
        }
    }
    __syncthreads();

    const int wave  = tid >> 5;
    const int lane  = tid & 31;
    const int half  = lane >> 4;     // which 16-lane half of the wave
    const int l16   = lane & 15;
    const int nbase = wave * 128;    // this wave's 128-column slice

    // bias per owned column (broadcast to every row of the C tiles)
    float bias[8];
    #pragma unroll
    for (int nt = 0; nt < 8; ++nt) bias[nt] = b1[nbase + nt * 16 + l16];

    #pragma unroll 1
    for (int mg = 0; mg < 2; ++mg) {          // 2 groups of 4 M-tiles = 128 rows
        v8f C[4][8];
        #pragma unroll
        for (int mm = 0; mm < 4; ++mm)
            #pragma unroll
            for (int nt = 0; nt < 8; ++nt) {
                v8f c;
                #pragma unroll
                for (int e = 0; e < 8; ++e) c[e] = bias[nt];
                C[mm][nt] = c;
            }

        #pragma unroll 1
        for (int k = 0; k < 16; ++k) {        // K = 512 in steps of 32
            // A fragments (contiguous k order, same as B):
            // lanes 0-15: K {0..15}; lanes 16-31: K {16..31}
            const int kb = k * 32 + half * 16;
            v16bf A[4];
            #pragma unroll
            for (int mm = 0; mm < 4; ++mm) {
                const __bf16* ar = &Atile[(mg * 64 + mm * 16 + l16) * LDS_STRIDE];
                v8bf lo = *(const v8bf*)(ar + kb);
                v8bf hi = *(const v8bf*)(ar + kb + 8);
                A[mm] = __builtin_shufflevector(lo, hi,
                        0,1,2,3,4,5,6,7,8,9,10,11,12,13,14,15);
            }
            #pragma unroll
            for (int nt = 0; nt < 8; ++nt) {
                // B fragment: lane owns column n = l16; halves split K 0..15/16..31
                // -> 16 contiguous bf16 = one aligned 32B load from W[n, k..]
                const __bf16* br = W + (size_t)(nbase + nt * 16 + l16) * INIT_DIM
                                     + k * 32 + half * 16;
                v16bf Bf = *(const v16bf*)br;
                #pragma unroll
                for (int mm = 0; mm < 4; ++mm)
                    C[mm][nt] = __builtin_amdgcn_wmma_f32_16x16x32_bf16(
                        false, A[mm], false, Bf, (short)0, C[mm][nt], false, false);
            }
        }

        // ReLU + reduce over the 64 rows of this M-group into column sums
        #pragma unroll
        for (int nt = 0; nt < 8; ++nt) {
            float s = 0.0f;
            #pragma unroll
            for (int mm = 0; mm < 4; ++mm) {
                v8f c = C[mm][nt];
                #pragma unroll
                for (int e = 0; e < 8; ++e) s += fmaxf(c[e], 0.0f);
            }
            __hip_atomic_fetch_add(&csum[nbase + nt * 16 + l16], s,
                                   __ATOMIC_RELAXED, __HIP_MEMORY_SCOPE_WORKGROUP);
        }
    }
    __syncthreads();

    for (int i = tid; i < HID; i += 256)
        __hip_atomic_fetch_add(&colsum[i], csum[i],
                               __ATOMIC_RELAXED, __HIP_MEMORY_SCOPE_AGENT);
}

// ---------------------------------------------------------------------------
// Kernel 3: pooled = relu(colsum/196608); logits = pooled @ Wc.T + bc
// ---------------------------------------------------------------------------
__global__ void k_final(const float* __restrict__ colsum,
                        const float* __restrict__ Wc,
                        const float* __restrict__ bc,
                        float* __restrict__ out) {
    const int o = threadIdx.x;            // 64 threads
    const float inv = 1.0f / (3.0f * 65536.0f);
    float acc = bc[o];
    const float4* w  = (const float4*)(Wc + o * HID);
    const float4* cs = (const float4*)colsum;
    for (int i = 0; i < HID / 4; ++i) {
        float4 c = cs[i];
        float4 ww = w[i];
        acc += fmaxf(c.x * inv, 0.0f) * ww.x
             + fmaxf(c.y * inv, 0.0f) * ww.y
             + fmaxf(c.z * inv, 0.0f) * ww.z
             + fmaxf(c.w * inv, 0.0f) * ww.w;
    }
    out[o] = acc;
}

// ---------------------------------------------------------------------------
// Input order (setup_inputs dict order):
// 0 feat | 1 all_feat_a 2 nbr_a 3 W1_a 4 b1_a | 5 all_feat_b 6 nbr_b 7 W1_b
// 8 b1_b | 9 all_feat_c 10 nbr_c 11 W1_c 12 b1_c | 13 Wc 14 bc
// ---------------------------------------------------------------------------
extern "C" void kernel_launch(void* const* d_in, const int* in_sizes, int n_in,
                              void* d_out, int out_size, void* d_ws, size_t ws_size,
                              hipStream_t stream) {
    const float* af_a = (const float*)d_in[1];
    const float* af_b = (const float*)d_in[5];
    const float* af_c = (const float*)d_in[9];
    const int*   nb_a = (const int*)d_in[2];
    const int*   nb_b = (const int*)d_in[6];
    const int*   nb_c = (const int*)d_in[10];
    const float* W1_a = (const float*)d_in[3];
    const float* W1_b = (const float*)d_in[7];
    const float* W1_c = (const float*)d_in[11];
    const float* b1_a = (const float*)d_in[4];
    const float* b1_b = (const float*)d_in[8];
    const float* b1_c = (const float*)d_in[12];
    const float* Wc   = (const float*)d_in[13];
    const float* bc   = (const float*)d_in[14];

    __bf16* wbf    = (__bf16*)d_ws;
    float*  colsum = (float*)((char*)d_ws + (size_t)3 * W_ELEMS * sizeof(__bf16));

    // 1) weight f32->bf16 + zero accumulator
    k_convert<<<dim3(W_ELEMS / 4 / 256, 3), 256, 0, stream>>>(W1_a, W1_b, W1_c,
                                                              wbf, colsum);
    // 2) fused gather-GEMM-ReLU-reduce
    k_main<<<dim3(N_NBR / ROWS_WG, 3), 256, 0, stream>>>(
        af_a, af_b, af_c, nb_a, nb_b, nb_c, b1_a, b1_b, b1_c, wbf, colsum);
    // 3) epilogue matvec
    k_final<<<1, OUTD, 0, stream>>>(colsum, Wc, bc, (float*)d_out);
}
